// HardOffsetAttention_76081050681599
// MI455X (gfx1250) — matrix operand order repaired and checked by reading, not compiled, not run on hardware
//
#include <hip/hip_runtime.h>

// ---------------------------------------------------------------------------
// out[b] = Z[b] @ M  where  Z[b,t,c] = sum_s attn[c>>9, t, s] * x[b, s, c&511]
//          M[h*512+k, n] = sum_d W_v[k, h*512+d] * W_o[h*512+d, n]
// Both matmuls run on v_wmma_f32_16x16x32_bf16 (f32 accumulate).
// ---------------------------------------------------------------------------

typedef __attribute__((ext_vector_type(16))) __bf16       v16bf;
typedef __attribute__((ext_vector_type(8)))  float        v8f;
typedef __attribute__((ext_vector_type(4)))  unsigned int v4u;

#define T_SEQ 33
#define DM    1024
#define TOK   512
#define ZP    1032                         // Z row pitch (bf16): 2064B, 16B-aligned, bank-skewed
#define SMEM_ZS_BYTES   (48 * ZP * 2)      // 99072
#define SMEM_ATTN_OFF   SMEM_ZS_BYTES
#define SMEM_BYTES      (SMEM_ZS_BYTES + 2 * T_SEQ * T_SEQ * 4)  // 107784

union FragBf { v16bf v; v4u u[2]; };

// ---------------- M precompute: Mt[n*1024 + k] (bf16, transposed) ----------
__global__ void __launch_bounds__(256) build_mt(const float* __restrict__ Wv,
                                                const float* __restrict__ Wo,
                                                __bf16* __restrict__ Mt) {
  const int tid = threadIdx.x;
  const int k0  = blockIdx.x * 4;          // 256 blocks cover k = 0..1023
  const int h   = k0 >> 9;
  const int kkb = k0 & (TOK - 1);
  const float* wo = Wo + (size_t)(h * 512) * DM;
  const float* wv = Wv + h * 512;

  float acc[4][4];
#pragma unroll
  for (int ki = 0; ki < 4; ++ki)
#pragma unroll
    for (int j = 0; j < 4; ++j) acc[ki][j] = 0.f;

  for (int d = 0; d < 512; ++d) {
    float w[4];
#pragma unroll
    for (int ki = 0; ki < 4; ++ki)                    // uniform -> scalar loads
      w[ki] = wv[(size_t)(kkb + ki) * DM + d];
    const float* worow = wo + (size_t)d * DM;
#pragma unroll
    for (int j = 0; j < 4; ++j) {
      const float o = worow[tid + 256 * j];           // coalesced
#pragma unroll
      for (int ki = 0; ki < 4; ++ki) acc[ki][j] += w[ki] * o;
    }
  }
#pragma unroll
  for (int j = 0; j < 4; ++j)
#pragma unroll
    for (int ki = 0; ki < 4; ++ki)
      Mt[(size_t)(tid + 256 * j) * DM + k0 + ki] = (__bf16)acc[ki][j];
}

// ---------------- fused attention kernel: one block per batch --------------
__global__ void __launch_bounds__(256) hard_attn_fused(const float* __restrict__ x,
                                                       const __bf16* __restrict__ Mt,
                                                       const float* __restrict__ attn,
                                                       float* __restrict__ out) {
  extern __shared__ char smem[];
  __bf16* Zs = (__bf16*)smem;                         // [48][ZP] bf16
  float*  at = (float*)(smem + SMEM_ATTN_OFF);        // [2*33*33] f32

  const int b    = blockIdx.x;
  const int tid  = threadIdx.x;
  const int lane = tid & 31;                          // wave32
  const int wave = tid >> 5;                          // 8 waves
  const int g    = lane >> 4;                         // lane group (K split)
  const int lm   = lane & 15;
  const float* xb = x + (size_t)b * T_SEQ * DM;

  // ---- Phase 1: stage attn in LDS, zero Z pad rows 33..47 ----
  for (int i = tid; i < 2 * T_SEQ * T_SEQ; i += 256) at[i] = attn[i];
  {
    unsigned int* zp = (unsigned int*)(Zs + 33 * ZP);
    for (int i = tid; i < (15 * ZP) / 2; i += 256) zp[i] = 0u;
  }
  __syncthreads();

  // ---- Build A-fragments of attn (16x32 bf16, ISA lane/K interleave) ----
  v16bf aZ[2][2];                                     // [m-tile][head]
#pragma unroll
  for (int m = 0; m < 2; ++m) {
#pragma unroll
    for (int h = 0; h < 2; ++h) {
      const int t = m * 16 + lm;                      // t <= 31
      v16bf a;
#pragma unroll
      for (int e = 0; e < 16; ++e) {
        const int K = 8 * g + e + ((e >= 8) ? 8 : 0); // K = s in 0..31
        a[e] = (__bf16)at[h * (T_SEQ * T_SEQ) + t * T_SEQ + K];
      }
      aZ[m][h] = a;
    }
  }

  // ---- Stage Z: Z[0..31, :] = attn_h(32x32-part) @ x  via WMMA ----
  for (int i = 0; i < 8; ++i) {
    const int ct = wave * 8 + i;                      // 64 col-tiles over 8 waves
    const int h  = ct >> 5;                           // head from channel
    const int c  = ct * 16 + lm;
    const int ck = c & (TOK - 1);
    v16bf bv;                                         // B frag: K=s, N=channel
#pragma unroll
    for (int e = 0; e < 16; ++e)
      bv[e] = (__bf16)xb[(size_t)(8 * g + e + ((e >= 8) ? 8 : 0)) * DM + ck];
#pragma unroll
    for (int m = 0; m < 2; ++m) {
      v8f acc = {};
      acc = __builtin_amdgcn_wmma_f32_16x16x32_bf16(false, aZ[m][h], false, bv,
                                                    (short)0, acc, false, false);
#pragma unroll
      for (int r = 0; r < 8; ++r)                     // D row = r + 8*g
        Zs[(m * 16 + 8 * g + r) * ZP + c] = (__bf16)acc[r];
    }
  }
  // attn row 32 is exactly delta(s==32): Z[32,:] = x[b,32,:]
  for (int i = tid; i < DM; i += 256)
    Zs[32 * ZP + i] = (__bf16)xb[(size_t)32 * DM + (i & (TOK - 1))];
  __syncthreads();

  // ---- Stage 2: out[b] = Z @ M,  3(m) x 4(n) accumulator tiles per wave ----
  float* ob = out + (size_t)b * T_SEQ * DM;
#pragma unroll
  for (int grp = 0; grp < 2; ++grp) {
    const int nt0 = wave * 8 + grp * 4;               // 64 n-tiles total
    v8f acc[3][4];
#pragma unroll
    for (int m = 0; m < 3; ++m)
#pragma unroll
      for (int j = 0; j < 4; ++j) acc[m][j] = (v8f){};

    for (int kt = 0; kt < 32; ++kt) {
      const int k0 = kt * 32;
      FragBf afr[3];
#pragma unroll
      for (int m = 0; m < 3; ++m) {                   // A from LDS Z (2x ds_read_b128)
        const char* ab = (const char*)(Zs + (m * 16 + lm) * ZP + k0 + 8 * g);
        afr[m].u[0] = *(const v4u*)ab;                // K = k0+8g .. +7
        afr[m].u[1] = *(const v4u*)(ab + 32);         // K = k0+8g+16 .. +23
      }
#pragma unroll
      for (int j = 0; j < 4; ++j) {                   // B from L2-resident Mt
        const int n = (nt0 + j) * 16 + lm;
        const char* bb = (const char*)(Mt + (size_t)n * DM + k0 + 8 * g);
        FragBf bfr;
        bfr.u[0] = *(const v4u*)bb;                   // K = k0+8g .. +7
        bfr.u[1] = *(const v4u*)(bb + 32);            // K = k0+8g+16 .. +23
#pragma unroll
        for (int m = 0; m < 3; ++m)
          acc[m][j] = __builtin_amdgcn_wmma_f32_16x16x32_bf16(
              false, afr[m].v, false, bfr.v, (short)0, acc[m][j], false, false);
      }
    }
    // epilogue: rows t = m*16 + 8g + r, guard t < 33
#pragma unroll
    for (int m = 0; m < 3; ++m) {
#pragma unroll
      for (int r = 0; r < 8; ++r) {
        const int t = m * 16 + 8 * g + r;
        if (t < T_SEQ) {
#pragma unroll
          for (int j = 0; j < 4; ++j)
            ob[(size_t)t * DM + (nt0 + j) * 16 + lm] = acc[m][j][r];
        }
      }
    }
  }
}

// ---------------------------------------------------------------------------
extern "C" void kernel_launch(void* const* d_in, const int* in_sizes, int n_in,
                              void* d_out, int out_size, void* d_ws, size_t ws_size,
                              hipStream_t stream) {
  (void)in_sizes; (void)n_in; (void)out_size; (void)ws_size;
  const float* x    = (const float*)d_in[0];
  const float* Wv   = (const float*)d_in[1];
  const float* Wo   = (const float*)d_in[2];
  const float* attn = (const float*)d_in[3];
  float* out = (float*)d_out;
  __bf16* Mt = (__bf16*)d_ws;                         // 1024*1024*2 B = 2 MB scratch

  // allow >64KB dynamic LDS (gfx1250 WGP has 320KB); harmless if already set
  (void)hipFuncSetAttribute(reinterpret_cast<const void*>(hard_attn_fused),
                            hipFuncAttributeMaxDynamicSharedMemorySize, SMEM_BYTES);

  build_mt<<<256, 256, 0, stream>>>(Wv, Wo, Mt);
  hard_attn_fused<<<4096, 256, SMEM_BYTES, stream>>>(x, Mt, attn, out);
}
